// MultiHeadSelfAttention_14568529068547
// MI455X (gfx1250) — matrix operand order repaired and checked by reading, not compiled, not run on hardware
//
#include <hip/hip_runtime.h>

typedef _Float16 half_t;
typedef __attribute__((ext_vector_type(16))) _Float16 v16h;
typedef __attribute__((ext_vector_type(8)))  _Float16 v8h;
typedef __attribute__((ext_vector_type(8)))  float    v8f;
typedef __attribute__((ext_vector_type(4)))  unsigned int u32x4;
typedef __attribute__((ext_vector_type(8)))  int i32x8;
typedef __attribute__((ext_vector_type(4)))  int i32x4;

#define D_MODEL  1024
#define NUM_HEADS 16
#define DK        64
#define SEQ       64
#define BATCH   1024
#define MROWS  (BATCH * SEQ)   // 65536

// workspace layout (units: half elements)
#define XH_OFF   0ull
#define WT_OFF   (XH_OFF + (unsigned long long)MROWS * D_MODEL)      // 4 transposed f16 weights
#define QKV_OFF  (WT_OFF + 4ull * D_MODEL * D_MODEL)                 // [b][h][3][64][64], V transposed
#define CTX_OFF  (QKV_OFF + 3ull * MROWS * D_MODEL)

#if defined(__has_builtin)
#if __has_builtin(__builtin_amdgcn_tensor_load_to_lds) && __has_builtin(__builtin_amdgcn_s_wait_tensorcnt)
#define HAVE_TDM 1
#endif
#endif
#ifndef HAVE_TDM
#define HAVE_TDM 0
#endif

__device__ __forceinline__ v16h cat8(v8h lo, v8h hi) {
    return __builtin_shufflevector(lo, hi, 0,1,2,3,4,5,6,7,8,9,10,11,12,13,14,15);
}

// ---------------- kernel 1: x f32 -> f16 ----------------
__global__ void k_convert_x(const float* __restrict__ x, half_t* __restrict__ xh) {
    size_t i = ((size_t)blockIdx.x * blockDim.x + threadIdx.x) * 8;
    float4 f0 = *(const float4*)(x + i);
    float4 f1 = *(const float4*)(x + i + 4);
    v8h h;
    h[0] = (half_t)f0.x; h[1] = (half_t)f0.y; h[2] = (half_t)f0.z; h[3] = (half_t)f0.w;
    h[4] = (half_t)f1.x; h[5] = (half_t)f1.y; h[6] = (half_t)f1.z; h[7] = (half_t)f1.w;
    *(v8h*)(xh + i) = h;
}

// ---------------- kernel 2: weights -> Wt f16 (Wt[n][k] = W[k][n]) ----------------
__global__ void k_transpose_w(const float* __restrict__ Wq, const float* __restrict__ Wk,
                              const float* __restrict__ Wv, const float* __restrict__ Wo,
                              half_t* __restrict__ wt) {
    const float* W = (blockIdx.z == 0) ? Wq : (blockIdx.z == 1) ? Wk : (blockIdx.z == 2) ? Wv : Wo;
    half_t* dst = wt + (size_t)blockIdx.z * D_MODEL * D_MODEL;
    __shared__ float tile[32][33];
    int n0 = blockIdx.x * 32, k0 = blockIdx.y * 32;
    int tx = threadIdx.x, ty = threadIdx.y;
    #pragma unroll
    for (int i = 0; i < 32; i += 8)
        tile[ty + i][tx] = W[(size_t)(k0 + ty + i) * D_MODEL + n0 + tx];
    __syncthreads();
    #pragma unroll
    for (int i = 0; i < 32; i += 8)
        dst[(size_t)(n0 + ty + i) * D_MODEL + k0 + tx] = (half_t)tile[tx][ty + i];
}

// ---------------- kernel 3: fused QKV projection (WMMA) ----------------
// grid (H, B), block 128 (4 waves, each a 32x32 quadrant of the 64x64 tile).
// One A fetch feeds all three weight matrices (12 WMMAs per K-step).
__global__ __launch_bounds__(128)
void k_qkv(const half_t* __restrict__ xh, const half_t* __restrict__ wt,
           const float* __restrict__ bq, const float* __restrict__ bk, const float* __restrict__ bv,
           half_t* __restrict__ qkv) {
    int h  = blockIdx.x;
    int bt = blockIdx.y;
    int w  = threadIdx.x >> 5;
    int wm = w >> 1, wn = w & 1;
    int L  = threadIdx.x & 31, lr = L & 15, lh = L >> 4;
    int m0 = bt * 64 + wm * 32;
    int n0 = h * 64 + wn * 32;
    v8f acc[3][2][2] = {};
    for (int k0 = 0; k0 < D_MODEL; k0 += 32) {
        v16h a[2];
        #pragma unroll
        for (int i = 0; i < 2; i++) {
            const half_t* ap = xh + (size_t)(m0 + i * 16 + lr) * D_MODEL + k0 + lh * 16;
            a[i] = cat8(*(const v8h*)ap, *(const v8h*)(ap + 8));
        }
        #pragma unroll
        for (int sel = 0; sel < 3; sel++) {
            const half_t* wsel = wt + (size_t)sel * D_MODEL * D_MODEL;
            v16h bm[2];
            #pragma unroll
            for (int j = 0; j < 2; j++)
                bm[j] = *(const v16h*)(wsel + (size_t)(n0 + j * 16 + lr) * D_MODEL + k0 + lh * 16);
            #pragma unroll
            for (int i = 0; i < 2; i++)
                #pragma unroll
                for (int j = 0; j < 2; j++)
                    acc[sel][i][j] = __builtin_amdgcn_wmma_f32_16x16x32_f16(
                        false, a[i], false, bm[j], (short)0, acc[sel][i][j], false, false);
        }
    }
    size_t base0 = ((size_t)bt * NUM_HEADS + h) * (3 * SEQ * DK);
    #pragma unroll
    for (int sel = 0; sel < 3; sel++) {
        const float* bias = (sel == 0) ? bq : (sel == 1) ? bk : bv;
        half_t* dst = qkv + base0 + (size_t)sel * SEQ * DK;
        #pragma unroll
        for (int i = 0; i < 2; i++)
            #pragma unroll
            for (int j = 0; j < 2; j++) {
                int d = wn * 32 + j * 16 + lr;          // 0..63 within head
                float bvv = bias[h * 64 + d];
                #pragma unroll
                for (int e = 0; e < 8; e++) {
                    int s = wm * 32 + i * 16 + lh * 8 + e;
                    float val = acc[sel][i][j][e] + bvv;
                    if (sel < 2) dst[(size_t)s * DK + d] = (half_t)val;
                    else         dst[(size_t)d * SEQ + s] = (half_t)val;  // V transposed
                }
            }
    }
}

// ---------------- kernel 4: attention per (b,h) ----------------
// grid (H, B), block 128; wave w owns rows 16w..16w+15 of the 64x64 score matrix.
// Q/K/Vt tile (24KB contiguous) staged into LDS by the Tensor Data Mover when available.
__global__ __launch_bounds__(128)
void k_attn(const half_t* __restrict__ qkv, const float* __restrict__ bias_table,
            half_t* __restrict__ ctx) {
    int h = blockIdx.x, b = blockIdx.y;
    __shared__ alignas(32) half_t QKVs[3 * SEQ * DK];   // [Q | K | Vt]
    __shared__ alignas(32) half_t Ps[SEQ * SEQ];
    size_t base = ((size_t)b * NUM_HEADS + h) * (3 * SEQ * DK);
    int tid = threadIdx.x;

#if HAVE_TDM
    if (tid < 32) {
        // 1-D tensor DMA: 3072 x 8-byte elements (24KB), global -> LDS (D# per ISA 08 §8.3/8.4)
        unsigned long long ga = (unsigned long long)(size_t)(qkv + base);
        unsigned lds_base = (unsigned)(size_t)(&QKVs[0]);
        u32x4 g0;
        g0.x = 1u;                                // count=1, user descriptor, no gather
        g0.y = lds_base;                          // lds_addr
        g0.z = (unsigned)ga;                      // global_addr[31:0]
        g0.w = (unsigned)(ga >> 32) | (2u << 30); // global_addr[56:32] | type=2
        i32x8 g1;
        g1[0] = 0x00030000;                       // workgroup_mask=0, data_size=3 (8B)
        g1[1] = (int)(3072u << 16);               // tensor_dim0.lo16 = 3072
        g1[2] = (int)(1u << 16);                  // tensor_dim0.hi=0, tensor_dim1.lo16=1
        g1[3] = (int)(3072u << 16);               // tensor_dim1.hi=0, tile_dim0=3072
        g1[4] = 1;                                // tile_dim1=1, tile_dim2=0
        g1[5] = 3072;                             // tensor_dim0_stride.lo32
        g1[6] = 0;                                // stride.hi, tensor_dim1_stride.lo
        g1[7] = 0;
        i32x4 gz4 = {0, 0, 0, 0};
        i32x8 gz8 = {0, 0, 0, 0, 0, 0, 0, 0};
        __builtin_amdgcn_tensor_load_to_lds(g0, g1, gz4, gz4, gz8, 0);
        __builtin_amdgcn_s_wait_tensorcnt(0);
    }
    __syncthreads();
#else
    #pragma unroll
    for (int i = 0; i < 12; i++) {
        int off = (i * 128 + tid) * 8;
        *(v8h*)(QKVs + off) = *(const v8h*)(qkv + base + off);
    }
    __syncthreads();
#endif

    const half_t* Qs = QKVs;
    const half_t* Ks = QKVs + SEQ * DK;
    const half_t* Vs = QKVs + 2 * SEQ * DK;     // Vt layout: [d][s]

    int w = tid >> 5, L = tid & 31, lr = L & 15, lh = L >> 4;

    // scores = Q @ K^T  (A rows = queries, B^T rows = keys, both row-major in LDS)
    v8f sc[4] = {};
    for (int k0 = 0; k0 < DK; k0 += 32) {
        const half_t* ap = Qs + (w * 16 + lr) * DK + k0 + lh * 16;
        v16h a = cat8(*(const v8h*)ap, *(const v8h*)(ap + 8));
        #pragma unroll
        for (int j = 0; j < 4; j++) {
            v16h bm = *(const v16h*)(Ks + (j * 16 + lr) * DK + k0 + lh * 16);
            sc[j] = __builtin_amdgcn_wmma_f32_16x16x32_f16(
                false, a, false, bm, (short)0, sc[j], false, false);
        }
    }

    // bias + softmax; row (w*16 + lh*8 + e) spans 16 lanes (same lh) x 4 accums
    const float scale = 0.125f;   // 1/sqrt(64)
    #pragma unroll
    for (int e = 0; e < 8; e++) {
        int qrow = w * 16 + lh * 8 + e;
        int rq = qrow >> 3, fq = qrow & 7;
        float pr[4];
        float mx = -3.0e38f;
        #pragma unroll
        for (int j = 0; j < 4; j++) {
            int kcol = j * 16 + lr;
            int rk = kcol >> 3, fk = kcol & 7;
            int dr = rq - rk; dr = dr < -7 ? -7 : (dr > 7 ? 7 : dr);
            int df = fq - fk; df = df < -7 ? -7 : (df > 7 ? 7 : df);
            pr[j] = sc[j][e] * scale +
                    bias_table[((dr + 7) * 15 + (df + 7)) * NUM_HEADS + h];
            mx = fmaxf(mx, pr[j]);
        }
        #pragma unroll
        for (int m = 8; m >= 1; m >>= 1) mx = fmaxf(mx, __shfl_xor(mx, m, 32));
        float sum = 0.f;
        #pragma unroll
        for (int j = 0; j < 4; j++) { pr[j] = __expf(pr[j] - mx); sum += pr[j]; }
        #pragma unroll
        for (int m = 8; m >= 1; m >>= 1) sum += __shfl_xor(sum, m, 32);
        float inv = 1.0f / sum;
        #pragma unroll
        for (int j = 0; j < 4; j++)
            Ps[qrow * SEQ + j * 16 + lr] = (half_t)(pr[j] * inv);
    }
    __syncthreads();

    // ctx = P @ V  (B^T rows = Vt rows, already in LDS)
    v8f ca[4] = {};
    for (int k0 = 0; k0 < SEQ; k0 += 32) {
        const half_t* ap = Ps + (w * 16 + lr) * SEQ + k0 + lh * 16;
        v16h a = cat8(*(const v8h*)ap, *(const v8h*)(ap + 8));
        #pragma unroll
        for (int j = 0; j < 4; j++) {
            v16h bm = *(const v16h*)(Vs + (j * 16 + lr) * SEQ + k0 + lh * 16);
            ca[j] = __builtin_amdgcn_wmma_f32_16x16x32_f16(
                false, a, false, bm, (short)0, ca[j], false, false);
        }
    }
    #pragma unroll
    for (int j = 0; j < 4; j++)
        #pragma unroll
        for (int e = 0; e < 8; e++) {
            int s = w * 16 + lh * 8 + e;
            int d = j * 16 + lr;
            ctx[((size_t)b * SEQ + s) * D_MODEL + h * DK + d] = (half_t)ca[j][e];
        }
}

// ---------------- kernel 5: out = ctx @ Wo + b_o (f32 out) ----------------
__global__ __launch_bounds__(128)
void k_outproj(const half_t* __restrict__ ctx, const half_t* __restrict__ wo,
               const float* __restrict__ bo, float* __restrict__ out) {
    int nt = blockIdx.x;
    int mt = blockIdx.y;
    int w  = threadIdx.x >> 5, wm = w >> 1, wn = w & 1;
    int L  = threadIdx.x & 31, lr = L & 15, lh = L >> 4;
    int m0 = mt * 64 + wm * 32;
    int n0 = nt * 64 + wn * 32;
    v8f acc[2][2] = {};
    for (int k0 = 0; k0 < D_MODEL; k0 += 32) {
        v16h a[2], bm[2];
        #pragma unroll
        for (int i = 0; i < 2; i++) {
            const half_t* ap = ctx + (size_t)(m0 + i * 16 + lr) * D_MODEL + k0 + lh * 16;
            a[i] = cat8(*(const v8h*)ap, *(const v8h*)(ap + 8));
        }
        #pragma unroll
        for (int j = 0; j < 2; j++) {
            const half_t* bp = wo + (size_t)(n0 + j * 16 + lr) * D_MODEL + k0 + lh * 16;
            bm[j] = *(const v16h*)bp;
        }
        #pragma unroll
        for (int i = 0; i < 2; i++)
            #pragma unroll
            for (int j = 0; j < 2; j++)
                acc[i][j] = __builtin_amdgcn_wmma_f32_16x16x32_f16(
                    false, a[i], false, bm[j], (short)0, acc[i][j], false, false);
    }
    #pragma unroll
    for (int i = 0; i < 2; i++)
        #pragma unroll
        for (int j = 0; j < 2; j++) {
            int col = n0 + j * 16 + lr;
            float bvv = bo[col];
            #pragma unroll
            for (int e = 0; e < 8; e++) {
                int row = m0 + i * 16 + lh * 8 + e;
                out[(size_t)row * D_MODEL + col] = acc[i][j][e] + bvv;
            }
        }
}

extern "C" void kernel_launch(void* const* d_in, const int* in_sizes, int n_in,
                              void* d_out, int out_size, void* d_ws, size_t ws_size,
                              hipStream_t stream) {
    const float* x  = (const float*)d_in[0];
    const float* Wq = (const float*)d_in[1];
    const float* bq = (const float*)d_in[2];
    const float* Wk = (const float*)d_in[3];
    const float* bk = (const float*)d_in[4];
    const float* Wv = (const float*)d_in[5];
    const float* bv = (const float*)d_in[6];
    const float* Wo = (const float*)d_in[7];
    const float* bo = (const float*)d_in[8];
    const float* bias_table = (const float*)d_in[9];

    half_t* ws  = (half_t*)d_ws;
    half_t* xh  = ws + XH_OFF;
    half_t* wt  = ws + WT_OFF;
    half_t* qkv = ws + QKV_OFF;
    half_t* ctx = ws + CTX_OFF;

    k_convert_x<<<(MROWS * (size_t)D_MODEL) / (256 * 8), 256, 0, stream>>>(x, xh);
    k_transpose_w<<<dim3(32, 32, 4), dim3(32, 8), 0, stream>>>(Wq, Wk, Wv, Wo, wt);
    k_qkv<<<dim3(NUM_HEADS, BATCH), 128, 0, stream>>>(xh, wt, bq, bk, bv, qkv);
    k_attn<<<dim3(NUM_HEADS, BATCH), 128, 0, stream>>>(qkv, bias_table, ctx);
    k_outproj<<<dim3(NUM_HEADS, BATCH), 128, 0, stream>>>(
        ctx, wt + 3ull * D_MODEL * D_MODEL, bo, (float*)d_out);
}
